// SwitchFeedForward_43903155700467
// MI455X (gfx1250) — matrix operand order repaired and checked by reading, compiled, not run on hardware
//
#include <hip/hip_runtime.h>
#include <math.h>

// Problem constants (B,S,D,F,E) = (4,2048,768,3072,8)
#define T_TOK 8192
#define DIM   768
#define FDIM  3072
#define NEXP  8
#define BM    128
#define BN    128
#define BK    32
#define TPAD  9216   // 72 * 128, covers worst-case per-expert 128-alignment padding

typedef __attribute__((ext_vector_type(16))) _Float16 v16h;
typedef __attribute__((ext_vector_type(8)))  float    v8f;

union Frag {
    uint4    q4[2];
    v16h     h;
};

// Low 32 bits of a flat LDS pointer == LDS byte offset (aperture rules, ISA 10.2)
__device__ __forceinline__ unsigned lds_off(const void* p) {
    return (unsigned)(size_t)p;
}

// 32 bytes per lane, global -> LDS via the CDNA5 async DMA path (ASYNCcnt).
// IOFFSET applies to both LDS and global addresses, so one pair + offset:16.
__device__ __forceinline__ void async_copy32(unsigned ldsByteOff, unsigned long long gaddr) {
    asm volatile("global_load_async_to_lds_b128 %0, %1, off\n\t"
                 "global_load_async_to_lds_b128 %0, %1, off offset:16"
                 :: "v"(ldsByteOff), "v"(gaddr) : "memory");
}

__device__ __forceinline__ void wait_async0() {
    asm volatile("s_wait_asynccnt 0x0" ::: "memory");
}

// ---------------------------------------------------------------- router
__global__ __launch_bounds__(256) void k_router(
    const float* __restrict__ x, const float* __restrict__ Ws,
    const float* __restrict__ bs, int* __restrict__ routes,
    float* __restrict__ pmax, int* __restrict__ counts) {
    int wave = threadIdx.x >> 5;
    int lane = threadIdx.x & 31;
    int t = blockIdx.x * 8 + wave;
    float acc[NEXP];
#pragma unroll
    for (int e = 0; e < NEXP; ++e) acc[e] = 0.f;
    const float* xr = x + (size_t)t * DIM;
    for (int i = 0; i < DIM / 32; ++i) {
        int d = i * 32 + lane;
        float xv = xr[d];
#pragma unroll
        for (int e = 0; e < NEXP; ++e) acc[e] += xv * Ws[d * NEXP + e];
    }
#pragma unroll
    for (int e = 0; e < NEXP; ++e)
        for (int off = 16; off >= 1; off >>= 1)
            acc[e] += __shfl_xor(acc[e], off, 32);
    if (lane == 0) {
        float l[NEXP];
        float m = -1e30f; int r = 0;
#pragma unroll
        for (int e = 0; e < NEXP; ++e) {
            l[e] = acc[e] + bs[e];
            if (l[e] > m) { m = l[e]; r = e; }
        }
        float s = 0.f;
#pragma unroll
        for (int e = 0; e < NEXP; ++e) s += expf(l[e] - m);
        pmax[t] = 1.0f / s;          // softmax max prob = exp(0)/sum
        routes[t] = r;
        atomicAdd(&counts[r], 1);
    }
}

// ---------------------------------------------------------------- bookkeeping
// meta layout (ints): [0:8) counts, [8:16) cursors, [16:25) aligned starts
__global__ void k_zero(int* __restrict__ gathered, int* __restrict__ meta) {
    int i = blockIdx.x * 256 + threadIdx.x;
    if (i < TPAD) gathered[i] = -1;
    if (i < 32) meta[i] = 0;
}

__global__ void k_offsets(int* __restrict__ meta) {
    if (threadIdx.x == 0) {
        int s = 0;
        for (int e = 0; e < NEXP; ++e) {
            meta[16 + e] = s;
            meta[8 + e]  = s;
            s += ((meta[e] + BM - 1) / BM) * BM;
        }
        meta[16 + NEXP] = s;
    }
}

__global__ void k_scatter(const int* __restrict__ routes, int* __restrict__ meta,
                          int* __restrict__ gathered) {
    int t = blockIdx.x * 256 + threadIdx.x;
    if (t < T_TOK) {
        int e = routes[t];
        int p = atomicAdd(&meta[8 + e], 1);
        gathered[p] = t;
    }
}

// ---------------------------------------------------------------- fp32 -> f16 converts
__global__ void k_cvt_x(const float* __restrict__ x, _Float16* __restrict__ xh) {
    size_t i = ((size_t)blockIdx.x * 256 + threadIdx.x) * 4;
    float4 v = *(const float4*)(x + i);
    struct H4 { _Float16 a, b, c, d; };
    H4 o = { (_Float16)v.x, (_Float16)v.y, (_Float16)v.z, (_Float16)v.w };
    *(H4*)(xh + i) = o;
}

// [E][R][C] fp32  ->  [E][C][R] f16 (K becomes contiguous for WMMA B fragments)
__global__ __launch_bounds__(256) void k_transpose(
    const float* __restrict__ in, _Float16* __restrict__ out, int R, int C) {
    __shared__ _Float16 tile[32][33];
    int e = blockIdx.z;
    int c0 = blockIdx.x * 32, r0 = blockIdx.y * 32;
    int tx = threadIdx.x & 31, ty = threadIdx.x >> 5;
    const float* src = in + (size_t)e * R * C;
    _Float16* dst = out + (size_t)e * R * C;
#pragma unroll
    for (int i = 0; i < 4; ++i)
        tile[ty + i * 8][tx] = (_Float16)src[(size_t)(r0 + ty + i * 8) * C + c0 + tx];
    __syncthreads();
#pragma unroll
    for (int i = 0; i < 4; ++i)
        dst[(size_t)(c0 + ty + i * 8) * R + r0 + tx] = tile[tx][ty + i * 8];
}

__device__ __forceinline__ int find_expert(const int* meta, int rowBase) {
    int e = 0;
#pragma unroll
    for (int i = NEXP - 1; i >= 0; --i)
        if (rowBase >= meta[16 + i]) { e = i; break; }
    return e;
}

// ---------------------------------------------------------------- GEMM1: h = gelu(X W1 + b1)
__global__ __launch_bounds__(256) void k_gemm1(
    const _Float16* __restrict__ xh, const _Float16* __restrict__ W1h,
    const float* __restrict__ b1, const int* __restrict__ gathered,
    const int* __restrict__ meta, _Float16* __restrict__ hbuf) {
    __shared__ unsigned ldsA[2 * BM * 16];   // double-buffered: 2 x 8 KB
    __shared__ unsigned ldsB[2 * BN * 16];

    int rowBase = blockIdx.y * BM;
    if (rowBase >= meta[16 + NEXP]) return;
    int e  = find_expert(meta, rowBase);
    int n0 = blockIdx.x * BN;

    int tid  = threadIdx.x;
    int wave = tid >> 5, lane = tid & 31;
    int hi16 = lane >> 4, lm = lane & 15;
    int wm = (wave >> 2) * 64;
    int wn = (wave & 3) * 32;

    int lr  = tid >> 1;         // 0..127 : LDS tile row owned by this thread
    int seg = tid & 1;          // which 32-byte half of the 64-byte row
    int tokenA = gathered[rowBase + lr];
    const _Float16* aSrc = xh + (size_t)(tokenA < 0 ? 0 : tokenA) * DIM + seg * 16;
    const _Float16* bSrc = W1h + ((size_t)e * FDIM + n0 + lr) * DIM + seg * 16;
    unsigned long long gA = (unsigned long long)(size_t)aSrc;
    unsigned long long gB = (unsigned long long)(size_t)bSrc;

    unsigned aOff0 = lds_off(&ldsA[lr * 16 + seg * 8]);
    unsigned aOff1 = aOff0 + BM * 16 * 4;
    unsigned bOff0 = lds_off(&ldsB[lr * 16 + seg * 8]);
    unsigned bOff1 = bOff0 + BN * 16 * 4;

    v8f zero8 = {0.f, 0.f, 0.f, 0.f, 0.f, 0.f, 0.f, 0.f};
    v8f acc[4][2];
#pragma unroll
    for (int mt = 0; mt < 4; ++mt)
#pragma unroll
        for (int nt = 0; nt < 2; ++nt) acc[mt][nt] = zero8;

    // stage K-tile 0 into buffer 0
    async_copy32(aOff0, gA);
    async_copy32(bOff0, gB);

#pragma unroll 2
    for (int kt = 0; kt < DIM / BK; ++kt) {
        wait_async0();                  // this wave's tile-kt DMA done
        __syncthreads();                // all waves done: tile-kt visible, tile-(kt-1) buffer free
        if (kt + 1 < DIM / BK) {        // overlap tile-(kt+1) DMA with compute
            unsigned la = ((kt + 1) & 1) ? aOff1 : aOff0;
            unsigned lb = ((kt + 1) & 1) ? bOff1 : bOff0;
            async_copy32(la, gA + 64ull * (kt + 1));
            async_copy32(lb, gB + 64ull * (kt + 1));
        }
        const uint4* A4 = (const uint4*)(ldsA + (size_t)(kt & 1) * (BM * 16));
        const uint4* B4 = (const uint4*)(ldsB + (size_t)(kt & 1) * (BN * 16));

        Frag af[4], bf[2];
#pragma unroll
        for (int mt = 0; mt < 4; ++mt) {      // A 16x32: words (j>>2)*8 + hi*4 + (j&3)
            int m = wm + mt * 16 + lm;
            af[mt].q4[0] = A4[m * 4 + hi16];
            af[mt].q4[1] = A4[m * 4 + 2 + hi16];
        }
#pragma unroll
        for (int nt = 0; nt < 2; ++nt) {      // B: same k order as A (words (j>>2)*8 + hi*4 + (j&3))
            int n = wn + nt * 16 + lm;
            bf[nt].q4[0] = B4[n * 4 + hi16];
            bf[nt].q4[1] = B4[n * 4 + 2 + hi16];
        }
#pragma unroll
        for (int mt = 0; mt < 4; ++mt)
#pragma unroll
            for (int nt = 0; nt < 2; ++nt)
                acc[mt][nt] = __builtin_amdgcn_wmma_f32_16x16x32_f16(
                    false, af[mt].h, false, bf[nt].h, (short)0, acc[mt][nt],
                    false, false);
    }

    float bb[2];
#pragma unroll
    for (int nt = 0; nt < 2; ++nt) bb[nt] = b1[e * FDIM + n0 + wn + nt * 16 + lm];
#pragma unroll
    for (int mt = 0; mt < 4; ++mt)
#pragma unroll
        for (int nt = 0; nt < 2; ++nt)
#pragma unroll
            for (int v = 0; v < 8; ++v) {
                int prow = rowBase + wm + mt * 16 + v + hi16 * 8;
                int fcol = n0 + wn + nt * 16 + lm;
                float hv = acc[mt][nt][v] + bb[nt];
                hv = 0.5f * hv * (1.0f + erff(hv * 0.70710678118f));  // exact gelu
                hbuf[(size_t)prow * FDIM + fcol] = (_Float16)hv;
            }
}

// ---------------------------------------------------------------- GEMM2: y = h W2 + b2, scatter*pmax
__global__ __launch_bounds__(256) void k_gemm2(
    const _Float16* __restrict__ hbuf, const _Float16* __restrict__ W2h,
    const float* __restrict__ b2, const int* __restrict__ gathered,
    const int* __restrict__ meta, const float* __restrict__ pmax,
    float* __restrict__ out) {
    __shared__ unsigned ldsA[2 * BM * 16];
    __shared__ unsigned ldsB[2 * BN * 16];

    int rowBase = blockIdx.y * BM;
    if (rowBase >= meta[16 + NEXP]) return;
    int e  = find_expert(meta, rowBase);
    int n0 = blockIdx.x * BN;                        // D tile

    int tid  = threadIdx.x;
    int wave = tid >> 5, lane = tid & 31;
    int hi16 = lane >> 4, lm = lane & 15;
    int wm = (wave >> 2) * 64;
    int wn = (wave & 3) * 32;

    int lr  = tid >> 1;
    int seg = tid & 1;
    const _Float16* aSrc = hbuf + (size_t)(rowBase + lr) * FDIM + seg * 16;
    const _Float16* bSrc = W2h + ((size_t)e * DIM + n0 + lr) * FDIM + seg * 16;
    unsigned long long gA = (unsigned long long)(size_t)aSrc;
    unsigned long long gB = (unsigned long long)(size_t)bSrc;

    unsigned aOff0 = lds_off(&ldsA[lr * 16 + seg * 8]);
    unsigned aOff1 = aOff0 + BM * 16 * 4;
    unsigned bOff0 = lds_off(&ldsB[lr * 16 + seg * 8]);
    unsigned bOff1 = bOff0 + BN * 16 * 4;

    v8f zero8 = {0.f, 0.f, 0.f, 0.f, 0.f, 0.f, 0.f, 0.f};
    v8f acc[4][2];
#pragma unroll
    for (int mt = 0; mt < 4; ++mt)
#pragma unroll
        for (int nt = 0; nt < 2; ++nt) acc[mt][nt] = zero8;

    async_copy32(aOff0, gA);
    async_copy32(bOff0, gB);

#pragma unroll 2
    for (int kt = 0; kt < FDIM / BK; ++kt) {
        wait_async0();
        __syncthreads();
        if (kt + 1 < FDIM / BK) {
            unsigned la = ((kt + 1) & 1) ? aOff1 : aOff0;
            unsigned lb = ((kt + 1) & 1) ? bOff1 : bOff0;
            async_copy32(la, gA + 64ull * (kt + 1));
            async_copy32(lb, gB + 64ull * (kt + 1));
        }
        const uint4* A4 = (const uint4*)(ldsA + (size_t)(kt & 1) * (BM * 16));
        const uint4* B4 = (const uint4*)(ldsB + (size_t)(kt & 1) * (BN * 16));

        Frag af[4], bf[2];
#pragma unroll
        for (int mt = 0; mt < 4; ++mt) {
            int m = wm + mt * 16 + lm;
            af[mt].q4[0] = A4[m * 4 + hi16];
            af[mt].q4[1] = A4[m * 4 + 2 + hi16];
        }
#pragma unroll
        for (int nt = 0; nt < 2; ++nt) {
            int n = wn + nt * 16 + lm;
            bf[nt].q4[0] = B4[n * 4 + hi16];
            bf[nt].q4[1] = B4[n * 4 + 2 + hi16];
        }
#pragma unroll
        for (int mt = 0; mt < 4; ++mt)
#pragma unroll
            for (int nt = 0; nt < 2; ++nt)
                acc[mt][nt] = __builtin_amdgcn_wmma_f32_16x16x32_f16(
                    false, af[mt].h, false, bf[nt].h, (short)0, acc[mt][nt],
                    false, false);
    }

#pragma unroll
    for (int mt = 0; mt < 4; ++mt)
#pragma unroll
        for (int nt = 0; nt < 2; ++nt)
#pragma unroll
            for (int v = 0; v < 8; ++v) {
                int prow = rowBase + wm + mt * 16 + v + hi16 * 8;
                int token = gathered[prow];
                if (token >= 0) {
                    int d = n0 + wn + nt * 16 + lm;
                    float y = acc[mt][nt][v] + b2[e * DIM + d];
                    out[(size_t)token * DIM + d] = y * pmax[token];
                }
            }
}

// ---------------------------------------------------------------- launch
extern "C" void kernel_launch(void* const* d_in, const int* in_sizes, int n_in,
                              void* d_out, int out_size, void* d_ws, size_t ws_size,
                              hipStream_t stream) {
    const float* x  = (const float*)d_in[0];
    const float* Ws = (const float*)d_in[1];
    const float* bs = (const float*)d_in[2];
    const float* W1 = (const float*)d_in[3];
    const float* b1 = (const float*)d_in[4];
    const float* W2 = (const float*)d_in[5];
    const float* b2 = (const float*)d_in[6];
    float* out = (float*)d_out;

    // workspace carve-up (~145 MB total)
    char* w = (char*)d_ws;
    int*      routes   = (int*)w;       w += (size_t)T_TOK * 4;
    float*    pmax     = (float*)w;     w += (size_t)T_TOK * 4;
    int*      meta     = (int*)w;       w += 256;
    int*      gathered = (int*)w;       w += (size_t)TPAD * 4;
    _Float16* xh       = (_Float16*)w;  w += (size_t)T_TOK * DIM * 2;
    _Float16* W1h      = (_Float16*)w;  w += (size_t)NEXP * DIM * FDIM * 2;
    _Float16* W2h      = (_Float16*)w;  w += (size_t)NEXP * DIM * FDIM * 2;
    _Float16* hbuf     = (_Float16*)w;  w += (size_t)TPAD * FDIM * 2;

    k_zero    <<<(TPAD + 255) / 256, 256, 0, stream>>>(gathered, meta);
    k_router  <<<T_TOK / 8, 256, 0, stream>>>(x, Ws, bs, routes, pmax, meta);
    k_offsets <<<1, 32, 0, stream>>>(meta);
    k_scatter <<<T_TOK / 256, 256, 0, stream>>>(routes, meta, gathered);
    k_cvt_x   <<<(T_TOK * DIM) / 1024, 256, 0, stream>>>(x, xh);
    k_transpose<<<dim3(FDIM / 32, DIM / 32, NEXP), 256, 0, stream>>>(W1, W1h, DIM, FDIM);
    k_transpose<<<dim3(DIM / 32, FDIM / 32, NEXP), 256, 0, stream>>>(W2, W2h, FDIM, DIM);
    k_gemm1   <<<dim3(FDIM / BN, TPAD / BM), 256, 0, stream>>>(xh, W1h, b1, gathered, meta, hbuf);
    k_gemm2   <<<dim3(DIM / BN, TPAD / BM), 256, 0, stream>>>(hbuf, W2h, b2, gathered, meta, pmax, out);
}
